// VGAE_encoder_24335284699606
// MI455X (gfx1250) — matrix operand order repaired and checked by reading, not compiled, not run on hardware
//
#include <hip/hip_runtime.h>
#include <hip/hip_bf16.h>

typedef __attribute__((ext_vector_type(16))) __bf16 v16bf;
typedef __attribute__((ext_vector_type(8)))  __bf16 bf16x8;
typedef __attribute__((ext_vector_type(8)))  float  v8f;

#define FEAT    128      // IN == H == 128
#define OUTC    64

// ---------------- utility kernels ----------------

__global__ void zero_f32(float* p, long long n) {
    long long i = (long long)blockIdx.x * blockDim.x + threadIdx.x;
    long long stride = (long long)gridDim.x * blockDim.x;
    for (; i < n; i += stride) p[i] = 0.0f;
}

__global__ void init_deg(float* deg, int n) {
    int i = blockIdx.x * blockDim.x + threadIdx.x;
    if (i < n) deg[i] = 1.0f;   // self-loop contribution
}

__global__ void accum_deg(const int* __restrict__ ei, float* deg, int E) {
    int e = blockIdx.x * blockDim.x + threadIdx.x;
    if (e < E) atomicAdd(&deg[ei[E + e]], 1.0f);  // dst row
}

__global__ void finalize_dinv(float* deg, int n) {
    int i = blockIdx.x * blockDim.x + threadIdx.x;
    if (i < n) deg[i] = rsqrtf(fmaxf(deg[i], 1.0f));
}

// x (f32, row-major [n]) -> bf16
__global__ void cvt_bf16(const float* __restrict__ x, __bf16* __restrict__ y, long long n) {
    long long i = (long long)blockIdx.x * blockDim.x + threadIdx.x;
    if (i < n) y[i] = (__bf16)x[i];
}

// W1 [K=128][Nc=128] row-major -> Wt [Nc][K] bf16 (transposed)
__global__ void pack_w1t(const float* __restrict__ W1, __bf16* __restrict__ Wt) {
    int i = blockIdx.x * blockDim.x + threadIdx.x;
    if (i >= FEAT * FEAT) return;
    int n = i / FEAT, k = i % FEAT;
    Wt[i] = (__bf16)W1[k * FEAT + n];
}

// [Wmu | Wsig] ([128][64] each) -> WcatT [Nc=128][K=128] bf16 (transposed)
__global__ void pack_wcat_t(const float* __restrict__ Wmu, const float* __restrict__ Wsig,
                            __bf16* __restrict__ Wt) {
    int i = blockIdx.x * blockDim.x + threadIdx.x;
    if (i >= FEAT * FEAT) return;
    int n = i / FEAT, k = i % FEAT;
    Wt[i] = (__bf16)((n < OUTC) ? Wmu[k * OUTC + n] : Wsig[k * OUTC + (n - OUTC)]);
}

// h_bf16 = relu(agg + b1)   (bf16 output feeds next GEMM directly)
__global__ void bias_relu_bf16(const float* __restrict__ agg, const float* __restrict__ b,
                               __bf16* __restrict__ h, long long n) {
    long long i = (long long)blockIdx.x * blockDim.x + threadIdx.x;
    if (i >= n) return;
    h[i] = (__bf16)fmaxf(agg[i] + b[(int)(i % FEAT)], 0.0f);
}

// ---------------- WMMA GEMM: C[M x 128] = A[M x 128] * Wt^T ----------------
// A: bf16 row-major [M][128]. Wt: bf16 [128][128] = W transposed ([col][k]).
// 256 threads = 8 waves; whole Wt staged in LDS (32 KB); each wave owns one
// 16-row stripe and all 8 column tiles (8 accumulators, A fragment reused 8x).

__global__ __launch_bounds__(256) void gemm_wmma_bf16(
        const __bf16* __restrict__ A, const __bf16* __restrict__ Wt,
        float* __restrict__ C, int M) {
    __shared__ __bf16 lds_w[FEAT * FEAT];   // 32 KB

    // cooperative Wt -> LDS (16-byte chunks)
    {
        const uint4* src = (const uint4*)Wt;
        uint4* dst = (uint4*)lds_w;
        const int nvec = FEAT * FEAT * 2 / 16;            // 2048
        for (int i = threadIdx.x; i < nvec; i += blockDim.x) dst[i] = src[i];
    }
    __syncthreads();

    const int wave = threadIdx.x >> 5;
    const int lane = threadIdx.x & 31;
    const int tileRow = blockIdx.x * 8 + wave;
    if (tileRow * 16 >= M) return;

    const int mrow = tileRow * 16 + (lane & 15);
    const int kgrp = lane >> 4;                 // 0: lanes 0-15, 1: lanes 16-31
    const __bf16* Arow = A + (size_t)mrow * FEAT;

    v8f acc[8] = {};
#pragma unroll
    for (int k0 = 0; k0 < FEAT; k0 += 32) {
        // A 16x32 fragment: elem e -> K = k0+kgrp*16+e
        const bf16x8 alo = *(const bf16x8*)(Arow + k0 + kgrp * 16);
        const bf16x8 ahi = *(const bf16x8*)(Arow + k0 + kgrp * 16 + 8);
        const v16bf a = __builtin_shufflevector(alo, ahi,
            0, 1, 2, 3, 4, 5, 6, 7, 8, 9, 10, 11, 12, 13, 14, 15);
#pragma unroll
        for (int t = 0; t < 8; ++t) {
            // B 32x16 fragment: elem e -> K = k0+kgrp*16+e, col = t*16+(lane&15)
            const int ncol = t * 16 + (lane & 15);
            const v16bf b = *(const v16bf*)(&lds_w[ncol * FEAT + k0 + kgrp * 16]);
            acc[t] = __builtin_amdgcn_wmma_f32_16x16x32_bf16(
                false, a, false, b, (short)0, acc[t], false, false);
        }
    }
    // C layout: VGPR r -> row tileRow*16 + kgrp*8 + r, col = t*16+(lane&15)
    const int rowBase = tileRow * 16 + kgrp * 8;
#pragma unroll
    for (int t = 0; t < 8; ++t) {
        const int ncol = t * 16 + (lane & 15);
#pragma unroll
        for (int r = 0; r < 8; ++r)
            C[(size_t)(rowBase + r) * FEAT + ncol] = acc[t][r];
    }
}

// ---------------- edge aggregation (one wave32 per edge, 128 features) ----------------

__global__ void aggregate128(const float* __restrict__ xw, const int* __restrict__ ei,
                             const float* __restrict__ dinv, float* __restrict__ out,
                             int E, int N) {
    unsigned wid = (blockIdx.x * blockDim.x + threadIdx.x) >> 5;
    int lane = threadIdx.x & 31;
    unsigned total = (unsigned)E + (unsigned)N;
    if (wid >= total) return;
    int s, d;
    if (wid < (unsigned)E) { s = ei[wid]; d = ei[E + wid]; }
    else                   { s = d = (int)(wid - (unsigned)E); }
    float nrm = dinv[s] * dinv[d];
    const float4 v = *(const float4*)(xw + (size_t)s * FEAT + lane * 4);
    float* o = out + (size_t)d * FEAT + lane * 4;
    atomicAdd(o + 0, v.x * nrm);
    atomicAdd(o + 1, v.y * nrm);
    atomicAdd(o + 2, v.z * nrm);
    atomicAdd(o + 3, v.w * nrm);
}

// Same but features [0,64) -> out_mu, [64,128) -> out_sig (d_out halves)
__global__ void aggregate_split(const float* __restrict__ xw, const int* __restrict__ ei,
                                const float* __restrict__ dinv,
                                float* __restrict__ out_mu, float* __restrict__ out_sig,
                                int E, int N) {
    unsigned wid = (blockIdx.x * blockDim.x + threadIdx.x) >> 5;
    int lane = threadIdx.x & 31;
    unsigned total = (unsigned)E + (unsigned)N;
    if (wid >= total) return;
    int s, d;
    if (wid < (unsigned)E) { s = ei[wid]; d = ei[E + wid]; }
    else                   { s = d = (int)(wid - (unsigned)E); }
    float nrm = dinv[s] * dinv[d];
    const float4 v = *(const float4*)(xw + (size_t)s * FEAT + lane * 4);
    float* o = (lane < 16) ? (out_mu  + (size_t)d * OUTC + lane * 4)
                           : (out_sig + (size_t)d * OUTC + (lane - 16) * 4);
    atomicAdd(o + 0, v.x * nrm);
    atomicAdd(o + 1, v.y * nrm);
    atomicAdd(o + 2, v.z * nrm);
    atomicAdd(o + 3, v.w * nrm);
}

// d_out: [N*64 mu][N*64 sig]; add respective biases
__global__ void bias_out(float* out, const float* __restrict__ bmu,
                         const float* __restrict__ bsig, int N) {
    long long i = (long long)blockIdx.x * blockDim.x + threadIdx.x;
    long long half = (long long)N * OUTC;
    if (i >= 2 * half) return;
    if (i < half) out[i] += bmu[(int)(i % OUTC)];
    else          out[i] += bsig[(int)((i - half) % OUTC)];
}

// ---------------- launch ----------------

extern "C" void kernel_launch(void* const* d_in, const int* in_sizes, int n_in,
                              void* d_out, int out_size, void* d_ws, size_t ws_size,
                              hipStream_t stream) {
    const float* x    = (const float*)d_in[0];
    const int*   ei   = (const int*)  d_in[1];
    const float* W1   = (const float*)d_in[2];
    const float* b1   = (const float*)d_in[3];
    const float* Wmu  = (const float*)d_in[4];
    const float* bmu  = (const float*)d_in[5];
    const float* Wsig = (const float*)d_in[6];
    const float* bsig = (const float*)d_in[7];
    float* out = (float*)d_out;

    const int N = in_sizes[0] / FEAT;       // 50000
    const int E = in_sizes[1] / 2;          // 800000
    const long long nf = (long long)N * FEAT;

    // workspace layout
    float*  dinv = (float*)d_ws;                           // N f32
    float*  xw   = dinv + N;                               // N*128 f32
    float*  agg  = xw + nf;                                // N*128 f32
    __bf16* abf  = (__bf16*)(agg + nf);                    // N*128 bf16 (x_bf, then h_bf)
    __bf16* Wt1  = abf + nf;                               // 128*128 bf16
    __bf16* WtC  = Wt1 + FEAT * FEAT;                      // 128*128 bf16

    const int B = 256;

    // 1) degree -> dinv
    init_deg<<<(N + B - 1) / B, B, 0, stream>>>(dinv, N);
    accum_deg<<<(E + B - 1) / B, B, 0, stream>>>(ei, dinv, E);
    finalize_dinv<<<(N + B - 1) / B, B, 0, stream>>>(dinv, N);

    // 2) pack transposed bf16 weights; convert x -> bf16
    pack_w1t<<<(FEAT * FEAT + B - 1) / B, B, 0, stream>>>(W1, Wt1);
    pack_wcat_t<<<(FEAT * FEAT + B - 1) / B, B, 0, stream>>>(Wmu, Wsig, WtC);
    cvt_bf16<<<(int)((nf + B - 1) / B), B, 0, stream>>>(x, abf, nf);

    // 3) xw = x @ W1  (WMMA, LDS-staged weights)
    {
        int rowTiles = N / 16;                  // 3125
        int blocks = (rowTiles + 7) / 8;
        gemm_wmma_bf16<<<blocks, B, 0, stream>>>(abf, Wt1, xw, N);
    }

    // 4) agg = Agg(xw); h_bf = relu(agg + b1)
    zero_f32<<<4096, B, 0, stream>>>(agg, nf);
    {
        long long threads = ((long long)E + N) * 32;
        aggregate128<<<(int)((threads + B - 1) / B), B, 0, stream>>>(xw, ei, dinv, agg, E, N);
    }
    bias_relu_bf16<<<(int)((nf + B - 1) / B), B, 0, stream>>>(agg, b1, abf, nf);

    // 5) xw2 = h @ [Wmu|Wsig]  (reuse xw buffer)
    {
        int rowTiles = N / 16;
        int blocks = (rowTiles + 7) / 8;
        gemm_wmma_bf16<<<blocks, B, 0, stream>>>(abf, WtC, xw, N);
    }

    // 6) d_out = Agg(xw2) split into mu | sig, then biases
    zero_f32<<<4096, B, 0, stream>>>(out, (long long)out_size);
    {
        long long threads = ((long long)E + N) * 32;
        aggregate_split<<<(int)((threads + B - 1) / B), B, 0, stream>>>(
            xw, ei, dinv, out, out + (size_t)N * OUTC, E, N);
    }
    bias_out<<<(int)((2LL * N * OUTC + B - 1) / B), B, 0, stream>>>(out, bmu, bsig, N);
}
